// MultiHeadAttention_74165495267740
// MI455X (gfx1250) — matrix operand order repaired and checked by reading, not compiled, not run on hardware
//
#include <hip/hip_runtime.h>
#include <hip/hip_bf16.h>

// ---------------------------------------------------------------------------
// MHA for MI455X (gfx1250): all matmuls via v_wmma_f32_16x16x32_f16 (wave32).
//   B=2, T=4096, D=1024, H=16, dk=64
// Pipeline:
//   1) proj_gemm<fp32 in, headsplit f16 out>  x3  -> Q,K,V  [B,H,T,64] f16
//   2) flash_attn (causal, online softmax)        -> attn   [B,T,D]    f16
//   3) proj_gemm<f16 in, fp32 flat out>           -> d_out  [B,T,D]    f32
// Both kernels use double-buffered LDS (1 barrier / K-step): global loads for
// step i+1 are issued into registers before computing step i, then stored to
// the alternate LDS buffer. Buffer i+1 was last READ at step i-1, which is
// ordered before the top-of-loop barrier of step i, so no extra barrier.
// ---------------------------------------------------------------------------

typedef _Float16 half_t;
typedef __attribute__((ext_vector_type(2)))  _Float16 h2;
typedef __attribute__((ext_vector_type(4)))  _Float16 v4h;
typedef __attribute__((ext_vector_type(16))) _Float16 v16h;
typedef __attribute__((ext_vector_type(8)))  float    v8f;

#define DMODEL 1024
#define TSEQ   4096
#define NHEAD  16
#define DK     64
#define BATCH  2
#define MROWS  (BATCH * TSEQ)   // 8192

__device__ __forceinline__ v8f wmma_f16(v16h a, v16h b, v8f c) {
  // D = A(16x32 f16) * B(32x16 f16) + C(16x16 f32)
  return __builtin_amdgcn_wmma_f32_16x16x32_f16(
      /*neg_a=*/false, a, /*neg_b=*/false, b,
      /*c_mod=*/(short)0, c, /*reuse_a=*/false, /*reuse_b=*/false);
}

// A-matrix fragment (16x32 f16, MxK). `row` -> 32 contiguous K halves of row
// M = lane&15. ISA 7.12.2: lanes 0-15 hold K 0-7 & 16-23; lanes 16-31 hold
// K 8-15 & 24-31 (two halves per VGPR).
__device__ __forceinline__ v16h frag_a_from_row(const half_t* row, int lane) {
  const int lo = (lane & 16) ? 8 : 0;
  v16h a;
#pragma unroll
  for (int i = 0; i < 8; ++i) a[i] = row[lo + i];
#pragma unroll
  for (int i = 0; i < 8; ++i) a[8 + i] = row[lo + 16 + i];
  return a;
}

// B-matrix fragment (32x16 f16, KxN). `col` -> 32 contiguous K halves of
// column N = lane&15. Same K order as A: lanes 0-15 K 0-7 & 16-23; lanes
// 16-31 K 8-15 & 24-31.
__device__ __forceinline__ v16h frag_b_from_col(const half_t* col, int lane) {
  const int base = (lane & 16) ? 8 : 0;
  v16h b;
#pragma unroll
  for (int i = 0; i < 8; ++i) b[i] = col[base + i];
#pragma unroll
  for (int i = 0; i < 8; ++i) b[8 + i] = col[base + 16 + i];
  return b;
}

union F4 { float4 v; float f[4]; };
union F4H { float4 v; half_t h[8]; };

// ---------------------------------------------------------------------------
// Projection GEMM: C[M=8192, N=1024] = A[M,1024] * W[1024,1024] + bias
// Block: 128 threads (4 waves). Block tile 128(M) x 64(N), K-step 32.
// Wave tile 32x64: 2 A-frags x 4 B-frags = 8 WMMA per K-step.
// ---------------------------------------------------------------------------
template <bool A_HALF, bool OUT_HEADSPLIT>
__global__ __launch_bounds__(128) void proj_gemm(
    const void* __restrict__ Aptr, const float* __restrict__ W,
    const float* __restrict__ bias, void* __restrict__ Out) {
  __shared__ half_t Xs[2][128][32];   // A tile, row-major (f16)
  __shared__ half_t Ws[2][64][32];    // W tile, TRANSPOSED: Ws[.][n][k]

  const int tid  = threadIdx.x;
  const int lane = tid & 31;
  const int wave = tid >> 5;
  const int ln   = lane & 15;
  const int hi8  = (lane >> 4) << 3;  // +8 row offset for upper half-wave

  const int n0 = blockIdx.x * 64;
  const int m0 = blockIdx.y * 128;

  // W staging: thread handles k-row pair (2p, 2p+1), 8 n-columns.
  const int wp = tid >> 3;            // 0..15  (k pair)
  const int wn = (tid & 7) * 8;       // 0..56  (n base) -> coalesced lanes

  v8f acc[2][4];
#pragma unroll
  for (int mt = 0; mt < 2; ++mt)
#pragma unroll
    for (int nt = 0; nt < 4; ++nt)
#pragma unroll
      for (int r = 0; r < 8; ++r) acc[mt][nt][r] = 0.0f;

  float4 xr[8];                 // A-tile staging regs (fp32: 8, f16: 4 used)
  float  wA[8], wB[8];          // W rows 2p / 2p+1

  auto load_regs = [&](int kb) {
    if (!A_HALF) {
      const float* X = (const float*)Aptr;
#pragma unroll
      for (int j = 0; j < 8; ++j) {
        const int idx = j * 128 + tid;          // 1024 float4 chunks
        const int r = idx >> 3, c = (idx & 7) * 4;
        xr[j] = *(const float4*)(X + (size_t)(m0 + r) * DMODEL + kb + c);
      }
    } else {
      const half_t* X = (const half_t*)Aptr;
#pragma unroll
      for (int j = 0; j < 4; ++j) {
        const int idx = j * 128 + tid;          // 512 8-half chunks
        const int r = idx >> 2, c = (idx & 3) * 8;
        xr[j] = *(const float4*)(X + (size_t)(m0 + r) * DMODEL + kb + c);
      }
    }
    const float* W0 = W + (size_t)(kb + 2 * wp) * DMODEL + n0 + wn;
    F4 a0, a1, b0, b1;
    a0.v = *(const float4*)(W0);
    a1.v = *(const float4*)(W0 + 4);
    b0.v = *(const float4*)(W0 + DMODEL);
    b1.v = *(const float4*)(W0 + DMODEL + 4);
#pragma unroll
    for (int i = 0; i < 4; ++i) {
      wA[i] = a0.f[i]; wA[4 + i] = a1.f[i];
      wB[i] = b0.f[i]; wB[4 + i] = b1.f[i];
    }
    // Prefetch next-next W tile (lowers to global_prefetch_b8).
    if (kb + 64 < DMODEL) __builtin_prefetch(W0 + 64 * DMODEL, 0, 1);
  };

  auto store_lds = [&](int buf) {
    if (!A_HALF) {
#pragma unroll
      for (int j = 0; j < 8; ++j) {
        const int idx = j * 128 + tid;
        const int r = idx >> 3, c = (idx & 7) * 4;
        v4h h;
        h[0] = (half_t)xr[j].x; h[1] = (half_t)xr[j].y;
        h[2] = (half_t)xr[j].z; h[3] = (half_t)xr[j].w;
        *(v4h*)&Xs[buf][r][c] = h;               // ds_store_b64
      }
    } else {
#pragma unroll
      for (int j = 0; j < 4; ++j) {
        const int idx = j * 128 + tid;
        const int r = idx >> 2, c = (idx & 3) * 8;
        *(float4*)&Xs[buf][r][c] = xr[j];        // ds_store_b128
      }
    }
#pragma unroll
    for (int i = 0; i < 8; ++i) {                // transposed, packed pairs
      h2 t; t[0] = (half_t)wA[i]; t[1] = (half_t)wB[i];
      *(h2*)&Ws[buf][wn + i][2 * wp] = t;        // ds_store_b32
    }
  };

  auto compute = [&](int buf) {
    const v16h a0 = frag_a_from_row(&Xs[buf][wave * 32 + ln][0], lane);
    const v16h a1 = frag_a_from_row(&Xs[buf][wave * 32 + 16 + ln][0], lane);
#pragma unroll
    for (int nt = 0; nt < 4; ++nt) {
      const v16h b = frag_b_from_col(&Ws[buf][nt * 16 + ln][0], lane);
      acc[0][nt] = wmma_f16(a0, b, acc[0][nt]);
      acc[1][nt] = wmma_f16(a1, b, acc[1][nt]);
    }
  };

  load_regs(0);
  store_lds(0);
  int buf = 0;
  for (int kb = 0; kb < DMODEL; kb += 32) {
    __syncthreads();                       // buf fully staged for all waves
    const bool more = (kb + 32 < DMODEL);
    if (more) load_regs(kb + 32);          // overlap global latency w/ WMMA
    compute(buf);
    if (more) store_lds(buf ^ 1);
    buf ^= 1;
  }

  // ---- epilogue: + bias, store ----
#pragma unroll
  for (int nt = 0; nt < 4; ++nt) {
    const int col = n0 + nt * 16 + ln;
    const float bv = bias[col];
#pragma unroll
    for (int mt = 0; mt < 2; ++mt)
#pragma unroll
      for (int r = 0; r < 8; ++r) {
        const int row = m0 + wave * 32 + mt * 16 + r + hi8;
        const float val = acc[mt][nt][r] + bv;
        if (OUT_HEADSPLIT) {
          half_t* O = (half_t*)Out;        // f16 [B,H,T,dk]
          const int b = row >> 12, t = row & (TSEQ - 1);
          const int h = col >> 6,  dk = col & (DK - 1);
          O[((size_t)(b * NHEAD + h) * TSEQ + t) * DK + dk] = (half_t)val;
        } else {
          ((float*)Out)[(size_t)row * DMODEL + col] = val;
        }
      }
  }
}

// ---------------------------------------------------------------------------
// Flash attention, causal. Grid: (T/64, B*H). Block: 128 threads (4 waves).
// Each wave owns 16 query rows; workgroup shares 32-key K/V LDS tiles
// (double buffered). Per key tile: 4 WMMA (S=QK^T) + 4 WMMA (O += P V).
// ---------------------------------------------------------------------------
__global__ __launch_bounds__(128) void flash_attn(
    const half_t* __restrict__ Q, const half_t* __restrict__ K,
    const half_t* __restrict__ V, half_t* __restrict__ Out) {
  __shared__ half_t Ksh[2][32][64];     // keys, row-major   [key][d]
  __shared__ half_t Vsh[2][64][32];     // values TRANSPOSED [d][key]
  __shared__ half_t Psh[4][16][32];     // per-wave P scratch (C->A relayout)

  const int tid  = threadIdx.x;
  const int lane = tid & 31;
  const int wave = tid >> 5;
  const int ln   = lane & 15;
  const int hi8  = (lane >> 4) << 3;

  const int qBase = blockIdx.x * 64;
  const int bh    = blockIdx.y;               // b*NHEAD + h
  const size_t head = (size_t)bh * TSEQ * DK;

  const int q0 = qBase + wave * 16;

  // Q fragments (A-layout) for this wave's 16 rows, K-dim 64 = 2 chunks.
  const half_t* qrow = Q + head + (size_t)(q0 + ln) * DK;
  const v16h aq0 = frag_a_from_row(qrow, lane);
  const v16h aq1 = frag_a_from_row(qrow + 32, lane);

  v8f acc[4];
  float mrow[8], lrow[8];
#pragma unroll
  for (int i = 0; i < 4; ++i)
#pragma unroll
    for (int r = 0; r < 8; ++r) acc[i][r] = 0.0f;
#pragma unroll
  for (int r = 0; r < 8; ++r) { mrow[r] = -3.0e38f; lrow[r] = 0.0f; }

  // V staging: thread handles key pair (2p, 2p+1), 8 d-columns (coalesced).
  const int vp = tid >> 3;            // 0..15 (key pair)
  const int vd = (tid & 7) * 8;       // 0..56 (d base)

  float4 kr[2];                       // K tile regs (2 x 8 halves)
  F4H    vr0, vr1;                    // V rows 2p / 2p+1 (8 halves each)

  auto load_regs = [&](int kBase) {
#pragma unroll
    for (int j = 0; j < 2; ++j) {
      const int idx = j * 128 + tid;  // 256 8-half chunks of 32x64 tile
      const int kk = idx >> 3, d = (idx & 7) * 8;
      kr[j] = *(const float4*)(K + head + (size_t)(kBase + kk) * DK + d);
    }
    vr0.v = *(const float4*)(V + head + (size_t)(kBase + 2 * vp) * DK + vd);
    vr1.v = *(const float4*)(V + head + (size_t)(kBase + 2 * vp + 1) * DK + vd);
    // Prefetch the tile after next (global_prefetch_b8).
    __builtin_prefetch(K + head + (size_t)(kBase + 32) * DK + vd, 0, 1);
  };

  auto store_lds = [&](int buf) {
#pragma unroll
    for (int j = 0; j < 2; ++j) {
      const int idx = j * 128 + tid;
      const int kk = idx >> 3, d = (idx & 7) * 8;
      *(float4*)&Ksh[buf][kk][d] = kr[j];        // ds_store_b128
    }
#pragma unroll
    for (int i = 0; i < 8; ++i) {                // transposed, packed pairs
      h2 t; t[0] = vr0.h[i]; t[1] = vr1.h[i];
      *(h2*)&Vsh[buf][vd + i][2 * vp] = t;       // ds_store_b32
    }
  };

  auto compute = [&](int buf, int kBase) {
    // ---- S = Q K^T : two 16x16 C tiles over 32 keys ----
    v8f s[2];
#pragma unroll
    for (int nt = 0; nt < 2; ++nt) {
      v8f sv;
#pragma unroll
      for (int r = 0; r < 8; ++r) sv[r] = 0.0f;
      sv = wmma_f16(aq0, frag_b_from_col(&Ksh[buf][nt * 16 + ln][0],  lane), sv);
      sv = wmma_f16(aq1, frag_b_from_col(&Ksh[buf][nt * 16 + ln][32], lane), sv);
      s[nt] = sv;
    }

    // ---- scale, causal mask, online softmax ----
    float e0[8], e1[8];
#pragma unroll
    for (int r = 0; r < 8; ++r) {
      const int row = q0 + r + hi8;
      const int c0 = kBase + ln, c1 = kBase + 16 + ln;
      float v0 = (c0 <= row) ? s[0][r] * 0.125f : -3.0e38f;  // 1/sqrt(64)
      float v1 = (c1 <= row) ? s[1][r] * 0.125f : -3.0e38f;
      float t = fmaxf(v0, v1);                 // row max across 16 lanes
      t = fmaxf(t, __shfl_xor(t, 1, 16));
      t = fmaxf(t, __shfl_xor(t, 2, 16));
      t = fmaxf(t, __shfl_xor(t, 4, 16));
      t = fmaxf(t, __shfl_xor(t, 8, 16));
      const float mn = fmaxf(mrow[r], t);
      const float p0 = __expf(v0 - mn);
      const float p1 = __expf(v1 - mn);
      float rs = p0 + p1;                      // row sum across 16 lanes
      rs += __shfl_xor(rs, 1, 16);
      rs += __shfl_xor(rs, 2, 16);
      rs += __shfl_xor(rs, 4, 16);
      rs += __shfl_xor(rs, 8, 16);
      const float fac = __expf(mrow[r] - mn);
      lrow[r] = lrow[r] * fac + rs;
      mrow[r] = mn;
      e0[r] = p0; e1[r] = p1;
#pragma unroll
      for (int dt = 0; dt < 4; ++dt) acc[dt][r] *= fac;
    }

    // ---- relayout P (C-layout) -> A-fragment via per-wave LDS scratch ----
#pragma unroll
    for (int r = 0; r < 8; ++r) {
      h2 t; t[0] = (half_t)e0[r]; t[1] = (half_t)0.0f;
      Psh[wave][r + hi8][ln]      = (half_t)e0[r];
      Psh[wave][r + hi8][16 + ln] = (half_t)e1[r];
    }
    const v16h ap = frag_a_from_row(&Psh[wave][ln][0], lane);

    // ---- O += P V : 4 dk subtiles ----
#pragma unroll
    for (int dt = 0; dt < 4; ++dt) {
      const v16h b = frag_b_from_col(&Vsh[buf][dt * 16 + ln][0], lane);
      acc[dt] = wmma_f16(ap, b, acc[dt]);
    }
  };

  const int nkt = (qBase + 64) >> 5;          // causal: keys <= qBase+63
  load_regs(0);
  store_lds(0);
  int buf = 0;
  for (int kt = 0; kt < nkt; ++kt) {
    __syncthreads();                          // buf staged for all waves
    const bool more = (kt + 1 < nkt);
    if (more) load_regs((kt + 1) << 5);
    compute(buf, kt << 5);
    if (more) store_lds(buf ^ 1);
    buf ^= 1;
  }

  // ---- normalize and store f16 [B,T,D] (heads merged) ----
  const int b = bh >> 4, h = bh & (NHEAD - 1);
#pragma unroll
  for (int dt = 0; dt < 4; ++dt)
#pragma unroll
    for (int r = 0; r < 8; ++r) {
      const int row = q0 + r + hi8;
      const int col = h * DK + dt * 16 + ln;
      Out[(size_t)(b * TSEQ + row) * DMODEL + col] =
          (half_t)(acc[dt][r] / lrow[r]);
    }
}

// ---------------------------------------------------------------------------
extern "C" void kernel_launch(void* const* d_in, const int* in_sizes, int n_in,
                              void* d_out, int out_size, void* d_ws,
                              size_t ws_size, hipStream_t stream) {
  (void)in_sizes; (void)n_in; (void)out_size; (void)ws_size;
  const float* x  = (const float*)d_in[0];
  // d_in[1] = causal mask: recomputed analytically, ignored.
  const float* Wq = (const float*)d_in[2];
  const float* bq = (const float*)d_in[3];
  const float* Wk = (const float*)d_in[4];
  const float* bk = (const float*)d_in[5];
  const float* Wv = (const float*)d_in[6];
  const float* bv = (const float*)d_in[7];
  const float* Wo = (const float*)d_in[8];
  const float* bo = (const float*)d_in[9];

  const size_t headElems = (size_t)BATCH * NHEAD * TSEQ * DK;  // 8M halves
  half_t* qh = (half_t*)d_ws;
  half_t* kh = qh + headElems;
  half_t* vh = kh + headElems;
  half_t* ah = vh + headElems;   // attention output, [B,T,D] f16

  const dim3 blk(128);
  const dim3 gProj(DMODEL / 64, MROWS / 128);   // (16, 64)

  proj_gemm<false, true><<<gProj, blk, 0, stream>>>((const void*)x, Wq, bq, qh);
  proj_gemm<false, true><<<gProj, blk, 0, stream>>>((const void*)x, Wk, bk, kh);
  proj_gemm<false, true><<<gProj, blk, 0, stream>>>((const void*)x, Wv, bv, vh);

  flash_attn<<<dim3(TSEQ / 64, BATCH * NHEAD), blk, 0, stream>>>(qh, kh, vh, ah);

  proj_gemm<true, false><<<gProj, blk, 0, stream>>>((const void*)ah, Wo, bo,
                                                    d_out);
}
